// TEM_56805237457573
// MI455X (gfx1250) — matrix operand order repaired and checked by reading, not compiled, not run on hardware
//
#include <hip/hip_runtime.h>
#include <hip/hip_bf16.h>

typedef _Float16 h16;
typedef __attribute__((ext_vector_type(16))) _Float16 v16h;
typedef __attribute__((ext_vector_type(8)))  _Float16 v8h;
typedef __attribute__((ext_vector_type(8)))  float    v8f;

#define XD 45
#define XCD 10
#define GDIM 20
#define PD 200
#define PK 224   // K padded to 7*32 for f16 WMMA
#define PN 208   // N padded to 13*16
#define NTI 13
#define KCH 7
#define PT 256   // row stride (halves) of d1/d2 staging
#define TB 256   // batch rows per block (8 waves x 32 rows)
#define KAPPA 0.8f
#define LAMDA 0.9999f
#define YITA 0.5f

// ---- WMMA fragment helpers (wave32, 16x16x32 f16) ----
__device__ __forceinline__ v16h pack16(v8h lo, v8h hi) {
  v16h r;
#pragma unroll
  for (int i = 0; i < 8; ++i) { r[i] = lo[i]; r[i + 8] = hi[i]; }
  return r;
}
// A matrix 16x32 f16: lane<16 holds row M=lane, K {0..7,16..23}; lane>=16 same row, K {8..15,24..31}
__device__ __forceinline__ v16h afrag(const h16* row, int k0, int h) {
  v8h lo = *(const v8h*)(row + k0 + 8 * h);
  v8h hi = *(const v8h*)(row + k0 + 16 + 8 * h);
  return pack16(lo, hi);
}
// B matrix 32x16 f16: lane = N (mod 16); lane<16 K {0..7,16..23}, lane>=16 K {8..15,24..31}
// (B stored N-major / K-contiguous, i.e. transposed source)
__device__ __forceinline__ v16h bfrag(const h16* row, int k0, int h) {
  v8h lo = *(const v8h*)(row + k0 + 8 * h);
  v8h hi = *(const v8h*)(row + k0 + 16 + 8 * h);
  return pack16(lo, hi);
}

__device__ __forceinline__ float act(float t) {
  t = (t > 0.f) ? t : 0.01f * t;               // leaky_relu
  return __builtin_amdgcn_fmed3f(t, -1.f, 1.f); // clip
}

// One wave retrieves its private 32 rows of p (32 x PK f16 slab in LDS), 5 iterations of
// p = clip(leaky_relu(p @ (M0 + kappa*I)), -1, 1).  kappa*p is folded into the diagonal of
// the staged matrix.  Two 16-row A tiles share every B fragment (halves LDS B traffic).
__device__ void wave_retrieve(h16* pw, const h16* m0t, int lane) {
  const int h = lane >> 4, ln = lane & 15;
#pragma unroll 1
  for (int it = 0; it < 5; ++it) {
    v16h a0[KCH], a1[KCH];
    const h16* ar0 = pw + ln * PK;
    const h16* ar1 = pw + (16 + ln) * PK;
#pragma unroll
    for (int kc = 0; kc < KCH; ++kc) {
      a0[kc] = afrag(ar0, kc * 32, h);
      a1[kc] = afrag(ar1, kc * 32, h);
    }
#pragma unroll 1
    for (int nt = 0; nt < NTI; ++nt) {
      v8f acc0 = {}, acc1 = {};
      const h16* brow = m0t + (nt * 16 + ln) * PK;
#pragma unroll
      for (int kc = 0; kc < KCH; ++kc) {
        v16h b = bfrag(brow, kc * 32, h);
        acc0 = __builtin_amdgcn_wmma_f32_16x16x32_f16(false, a0[kc], false, b,
                                                      (short)0, acc0, false, false);
        acc1 = __builtin_amdgcn_wmma_f32_16x16x32_f16(false, a1[kc], false, b,
                                                      (short)0, acc1, false, false);
      }
      const int col = nt * 16 + ln;
      const bool live = (col < PD);
#pragma unroll
      for (int v = 0; v < 8; ++v) {
        const int row = v + 8 * h;
        float t0 = act(acc0[v]);
        float t1 = act(acc1[v]);
        pw[row * PK + col]        = live ? (h16)t0 : (h16)0.f;
        pw[(16 + row) * PK + col] = live ? (h16)t1 : (h16)0.f;
      }
    }
  }
}

// small-weight LDS offsets (floats)
#define O_WC   0
#define O_W1I  450
#define O_B1I  1250
#define O_W2I  1290
#define O_B2I  2090
#define O_W1G  2110
#define O_B1G  2910
#define O_W2G  2950
#define O_B2G  3750
#define O_WSP  3770
#define O_BSP  4220
#define NWTOT  4265

__global__ __launch_bounds__(256)
void tem_main_kernel(const float* __restrict__ x, const float* __restrict__ g,
                     const float* __restrict__ M,
                     const float* __restrict__ Wc,  const float* __restrict__ W1i,
                     const float* __restrict__ b1i, const float* __restrict__ W2i,
                     const float* __restrict__ b2i, const float* __restrict__ W1g,
                     const float* __restrict__ b1g, const float* __restrict__ W2g,
                     const float* __restrict__ b2g, const float* __restrict__ Wsp,
                     const float* __restrict__ bsp,
                     float* __restrict__ o_ginf, float* __restrict__ o_xinf,
                     float* __restrict__ o_g, float* __restrict__ o_x,
                     h16* __restrict__ d1g, h16* __restrict__ d2g) {
  __shared__ h16  sM0T[PN * PK];        // 93 KB: (M0 + kappa*I)^T, f16, zero padded
  __shared__ h16  sP[TB * PK];          // 114 KB: p rows (wave w owns rows [32w,32w+32))
  __shared__ float sXC[TB * XCD];       // xc per row
  __shared__ float sG2[TB * GDIM];      // g2 per row
  __shared__ float sW[NWTOT];           // small weights

  const int tid = threadIdx.x;
  const int lane = tid & 31;
  const int wave = tid >> 5;
  const long long gr0 = (long long)blockIdx.x * TB;

  // ---- phase 0: stage (M0 + kappa*I)^T (f16) + small weights into LDS ----
  for (int idx = tid; idx < PN * PK; idx += 256) {
    int n = idx / PK, k = idx - n * PK;
    float v = 0.f;
    if (n < PD && k < PD) v = M[k * PD + n] + ((k == n) ? KAPPA : 0.f);
    sM0T[idx] = (h16)v;
  }
  for (int i = tid; i < 450; i += 256) sW[O_WC + i]  = Wc[i];
  for (int i = tid; i < 800; i += 256) sW[O_W1I + i] = W1i[i];
  for (int i = tid; i < 40;  i += 256) sW[O_B1I + i] = b1i[i];
  for (int i = tid; i < 800; i += 256) sW[O_W2I + i] = W2i[i];
  for (int i = tid; i < 20;  i += 256) sW[O_B2I + i] = b2i[i];
  for (int i = tid; i < 800; i += 256) sW[O_W1G + i] = W1g[i];
  for (int i = tid; i < 40;  i += 256) sW[O_B1G + i] = b1g[i];
  for (int i = tid; i < 800; i += 256) sW[O_W2G + i] = W2g[i];
  for (int i = tid; i < 20;  i += 256) sW[O_B2G + i] = b2g[i];
  for (int i = tid; i < 450; i += 256) sW[O_WSP + i] = Wsp[i];
  for (int i = tid; i < 45;  i += 256) sW[O_BSP + i] = bsp[i];
  __syncthreads();

  // ---- phase 1: per-row xc = x@W_comp, g2 = tanh(MLP_gg(g)) (one row per thread) ----
  {
    const float* xr = x + (gr0 + tid) * XD;
    float xc[XCD];
#pragma unroll
    for (int c = 0; c < XCD; ++c) xc[c] = 0.f;
    for (int i = 0; i < XD; ++i) {
      float xv = xr[i];
#pragma unroll
      for (int c = 0; c < XCD; ++c) xc[c] += xv * sW[O_WC + i * XCD + c];
    }
#pragma unroll
    for (int c = 0; c < XCD; ++c) sXC[tid * XCD + c] = xc[c];

    const float* grw = g + (gr0 + tid) * GDIM;
    float gv[GDIM];
#pragma unroll
    for (int i = 0; i < GDIM; ++i) gv[i] = grw[i];
    float h1[2 * GDIM];
#pragma unroll
    for (int j = 0; j < 2 * GDIM; ++j) h1[j] = sW[O_B1G + j];
    for (int i = 0; i < GDIM; ++i) {
#pragma unroll
      for (int j = 0; j < 2 * GDIM; ++j) h1[j] += gv[i] * sW[O_W1G + i * 2 * GDIM + j];
    }
#pragma unroll
    for (int j = 0; j < 2 * GDIM; ++j) h1[j] = (h1[j] > 0.f) ? h1[j] : (__expf(h1[j]) - 1.f);
    for (int o = 0; o < GDIM; ++o) {
      float s = sW[O_B2G + o];
      for (int j = 0; j < 2 * GDIM; ++j) s += h1[j] * sW[O_W2G + j * GDIM + o];
      sG2[tid * GDIM + o] = tanhf(s);
    }
  }
  __syncthreads();

  // ---- phase 1b: write x_ / g_ outputs (coalesced), init p slab with x_ cue ----
  for (int idx = tid; idx < TB * PD; idx += 256) {
    int r = idx / PD, c = idx - r * PD;
    long long go = (gr0 + r) * (long long)PD + c;
    o_x[go] = sXC[r * XCD + (c % XCD)];
    o_g[go] = sG2[r * GDIM + (c / XCD)];
  }
  {
    h16* pw = sP + wave * 32 * PK;
    for (int idx = lane; idx < 32 * PK; idx += 32) {
      int r = idx / PK, k = idx - r * PK;
      pw[idx] = (k < PD) ? (h16)sXC[(wave * 32 + r) * XCD + (k % XCD)] : (h16)0.f;
    }
    // ---- phase 2: retrieval #1 (px from x_) ----
    wave_retrieve(pw, sM0T, lane);
  }
  __syncthreads();

  // ---- phase 3: g_inf = clip(MLP_ie(px @ W_repeat^T), -1, 1) ----
  {
    const h16* prow = sP + tid * PK;
    float gsum[GDIM];
#pragma unroll
    for (int gi = 0; gi < GDIM; ++gi) {
      float s = 0.f;
#pragma unroll
      for (int c = 0; c < XCD; ++c) s += (float)prow[gi * XCD + c];
      gsum[gi] = s;
    }
    float h1[2 * GDIM];
#pragma unroll
    for (int j = 0; j < 2 * GDIM; ++j) h1[j] = sW[O_B1I + j];
    for (int i = 0; i < GDIM; ++i) {
#pragma unroll
      for (int j = 0; j < 2 * GDIM; ++j) h1[j] += gsum[i] * sW[O_W1I + i * 2 * GDIM + j];
    }
#pragma unroll
    for (int j = 0; j < 2 * GDIM; ++j) h1[j] = (h1[j] > 0.f) ? h1[j] : (__expf(h1[j]) - 1.f);
    float* gout = o_ginf + (gr0 + tid) * GDIM;
    for (int o = 0; o < GDIM; ++o) {
      float s = sW[O_B2I + o];
      for (int j = 0; j < 2 * GDIM; ++j) s += h1[j] * sW[O_W2I + j * GDIM + o];
      gout[o] = fminf(1.f, fmaxf(-1.f, s));
    }
  }
  __syncthreads();

  // ---- phase 4: re-init p slab with g_ cue, retrieval #2 ----
  {
    h16* pw = sP + wave * 32 * PK;
    for (int idx = lane; idx < 32 * PK; idx += 32) {
      int r = idx / PK, k = idx - r * PK;
      pw[idx] = (k < PD) ? (h16)sG2[(wave * 32 + r) * GDIM + (k / XCD)] : (h16)0.f;
    }
    wave_retrieve(pw, sM0T, lane);
  }
  __syncthreads();

  // ---- phase 5: x_inf = (p @ W_tile^T) @ W_sp + b_sp ----
  {
    const h16* prow = sP + tid * PK;
    float xsum[XCD];
#pragma unroll
    for (int c = 0; c < XCD; ++c) {
      float s = 0.f;
#pragma unroll
      for (int gi = 0; gi < GDIM; ++gi) s += (float)prow[gi * XCD + c];
      xsum[c] = s;
    }
    float* xout = o_xinf + (gr0 + tid) * XD;
    for (int o = 0; o < XD; ++o) {
      float s = sW[O_BSP + o];
#pragma unroll
      for (int c = 0; c < XCD; ++c) s += xsum[c] * sW[O_WSP + c * XD + o];
      xout[o] = s;
    }
  }

  // ---- phase 6: stage d1 = p_inf - p, d2 = p_inf + p (f16, stride PT, zero padded) ----
  for (int idx = tid; idx < TB * PT; idx += 256) {
    int r = idx >> 8, c = idx & (PT - 1);
    float d1 = 0.f, d2 = 0.f;
    if (c < PD) {
      float gx = sG2[r * GDIM + (c / XCD)] * sXC[r * XCD + (c % XCD)];
      float pi = (gx > 0.f) ? gx : 0.01f * gx;
      pi = __builtin_amdgcn_fmed3f(pi, -1.f, 1.f);
      float pv = (float)sP[r * PK + c];
      d1 = pi - pv;
      d2 = pi + pv;
    }
    long long go = (gr0 + r) * (long long)PT + c;
    d1g[go] = (h16)d1;
    d2g[go] = (h16)d2;
  }
}

// hebb partial: hws[m][n] += sum_k d1[k][m] * d2[k][n], K split over blockIdx.z
__global__ __launch_bounds__(256)
void tem_hebb_kernel(const h16* __restrict__ d1g, const h16* __restrict__ d2g,
                     float* __restrict__ hws, int chunksPerBlock) {
  __shared__ h16 t1[64 * 40];   // d1 chunk transposed: t1[m][k], stride 40
  __shared__ h16 t2[64 * 40];   // d2 chunk transposed: t2[n][k]
  const int tid = threadIdx.x, lane = tid & 31, wave = tid >> 5;
  const int h = lane >> 4, ln = lane & 15;
  const int mbase = blockIdx.x * 64, nbase = blockIdx.y * 64;
  const long long kstart = (long long)blockIdx.z * chunksPerBlock;

  const int tA = wave, tB = wave + 8;            // 16 tiles (4x4) over 8 waves
  const int mA = tA >> 2, nA = tA & 3;
  const int mB = tB >> 2, nB = tB & 3;
  v8f accA = {}, accB = {};

  // each thread stages one v8h (global_load_b128) per buffer per chunk
  const int skk = tid >> 3;        // 0..31 : k within chunk
  const int smg = tid & 7;         // 0..7  : group of 8 columns
  for (int c = 0; c < chunksPerBlock; ++c) {
    const long long k0 = (kstart + c) * 32;
    {
      v8h a = *(const v8h*)(d1g + (k0 + skk) * PT + mbase + smg * 8);
      v8h b = *(const v8h*)(d2g + (k0 + skk) * PT + nbase + smg * 8);
#pragma unroll
      for (int i = 0; i < 8; ++i) {
        t1[(smg * 8 + i) * 40 + skk] = a[i];
        t2[(smg * 8 + i) * 40 + skk] = b[i];
      }
    }
    __syncthreads();
    {
      v16h a = afrag(t1 + (mA * 16 + ln) * 40, 0, h);
      v16h b = bfrag(t2 + (nA * 16 + ln) * 40, 0, h);
      accA = __builtin_amdgcn_wmma_f32_16x16x32_f16(false, a, false, b, (short)0, accA, false, false);
      a = afrag(t1 + (mB * 16 + ln) * 40, 0, h);
      b = bfrag(t2 + (nB * 16 + ln) * 40, 0, h);
      accB = __builtin_amdgcn_wmma_f32_16x16x32_f16(false, a, false, b, (short)0, accB, false, false);
    }
    __syncthreads();
  }
#pragma unroll
  for (int v = 0; v < 8; ++v) {
    int gmA = mbase + mA * 16 + v + 8 * h, gnA = nbase + nA * 16 + ln;
    int gmB = mbase + mB * 16 + v + 8 * h, gnB = nbase + nB * 16 + ln;
    atomicAdd(&hws[gmA * PT + gnA], accA[v]);
    atomicAdd(&hws[gmB * PT + gnB], accB[v]);
  }
}

__global__ void tem_finalize_kernel(const float* __restrict__ M,
                                    const float* __restrict__ hws,
                                    float* __restrict__ outM, float invB) {
  int idx = blockIdx.x * blockDim.x + threadIdx.x;
  if (idx < PD * PD) {
    int i = idx / PD, j = idx - i * PD;
    outM[idx] = LAMDA * M[idx] + YITA * invB * hws[i * PT + j];
  }
}

extern "C" void kernel_launch(void* const* d_in, const int* in_sizes, int n_in,
                              void* d_out, int out_size, void* d_ws, size_t ws_size,
                              hipStream_t stream) {
  (void)n_in; (void)out_size; (void)ws_size;
  const float* x   = (const float*)d_in[0];
  const float* g   = (const float*)d_in[1];
  const float* M   = (const float*)d_in[2];
  const float* Wc  = (const float*)d_in[3];
  const float* W1i = (const float*)d_in[4];
  const float* b1i = (const float*)d_in[5];
  const float* W2i = (const float*)d_in[6];
  const float* b2i = (const float*)d_in[7];
  const float* W1g = (const float*)d_in[8];
  const float* b1g = (const float*)d_in[9];
  const float* W2g = (const float*)d_in[10];
  const float* b2g = (const float*)d_in[11];
  const float* Wsp = (const float*)d_in[12];
  const float* bsp = (const float*)d_in[13];

  const long long B = in_sizes[0] / XD;   // 131072
  float* out = (float*)d_out;
  float* o_ginf = out;
  float* o_xinf = o_ginf + B * GDIM;
  float* o_g    = o_xinf + B * XD;
  float* o_x    = o_g    + B * PD;
  float* o_M    = o_x    + B * PD;

  char* ws = (char*)d_ws;
  h16*   d1g = (h16*)ws;
  h16*   d2g = (h16*)(ws + (size_t)B * PT * sizeof(h16));
  float* hws = (float*)(ws + 2 * (size_t)B * PT * sizeof(h16));

  hipMemsetAsync(hws, 0, (size_t)PT * PT * sizeof(float), stream);

  dim3 mgrid((unsigned)(B / TB));
  tem_main_kernel<<<mgrid, 256, 0, stream>>>(x, g, M, Wc, W1i, b1i, W2i, b2i,
                                             W1g, b1g, W2g, b2g, Wsp, bsp,
                                             o_ginf, o_xinf, o_g, o_x, d1g, d2g);

  const int KSPLIT = 64;
  int chunksPerBlock = (int)((B / 32) / KSPLIT);   // 64 for B=131072
  tem_hebb_kernel<<<dim3(4, 4, KSPLIT), 256, 0, stream>>>(d1g, d2g, hws, chunksPerBlock);

  tem_finalize_kernel<<<(PD * PD + 255) / 256, 256, 0, stream>>>(M, hws, o_M, 1.f / (float)B);
}
